// AttentionLayer_42760694399404
// MI455X (gfx1250) — matrix operand order repaired and checked by reading, not compiled, not run on hardware
//
#include <hip/hip_runtime.h>

// ---------------------------------------------------------------------------
// B=16, S=2048, C=512, fp32 in/out.  bf16 WMMA everywhere, fused flash
// attention (score matrix never touches HBM).  Workspace:
//   [0,   32M) : K  = bf16(x' @ W^T + b)   [B*S, C]
//   [32M, 64M) : Q  = bf16(tanh(x'))       [B*S, C]
//   [64M, 96M) : V  = bf16(x)              [B*S, C]
//   [96M, ...) : Wb = bf16(W)              [C, C]
// ---------------------------------------------------------------------------

#define BATCH 16
#define SEQ   2048
#define CH    512
#define BM    64      // query rows per workgroup (4 waves x 16 rows)
#define BN    64      // key rows per tile

typedef __bf16 bf16_t;
typedef bf16_t v8bf  __attribute__((ext_vector_type(8)));
typedef bf16_t v16bf __attribute__((ext_vector_type(16)));
typedef float  v8f   __attribute__((ext_vector_type(8)));

// ---------------------------------------------------------------------------
// CDNA5 async global->LDS copy (ASYNCcnt path), 16 bytes per lane.
// ---------------------------------------------------------------------------
__device__ __forceinline__ void async_copy_b128(void* lds_dst, const void* gsrc) {
  unsigned int ldsa = (unsigned int)(uintptr_t)lds_dst;  // flat->LDS: low 32 bits
  asm volatile("global_load_async_to_lds_b128 %0, %1, off"
               :: "v"(ldsa), "v"(gsrc)
               : "memory");
}
__device__ __forceinline__ void wait_async0() {
  asm volatile("s_wait_asynccnt 0" ::: "memory");
}

// ---------------------------------------------------------------------------
// WMMA wrapper: D = A(16x32 bf16) x B(32x16 bf16) + C(16x16 f32)
// ---------------------------------------------------------------------------
__device__ __forceinline__ v8f wmma_bf16(v16bf a, v16bf b, v8f c) {
  return __builtin_amdgcn_wmma_f32_16x16x32_bf16(
      /*neg_a=*/false, a, /*neg_b=*/false, b,
      /*c_mod=*/(short)0, c, /*reuse_a=*/false, /*reuse_b=*/false);
}

// A-matrix 16x32 bf16 fragment (ISA 7.12.2): lane m=lane&15, half g=lane>>4.
// elems 0..7  <- Mat[m][k + g*8 + 0..7] ; elems 8..15 <- Mat[m][k+16+g*8+0..7]
__device__ __forceinline__ v16bf load_a_frag(const bf16_t* p0, int stride, int lane) {
  const int m = lane & 15, g = lane >> 4;
  const bf16_t* p = p0 + (size_t)m * stride + g * 8;
  v8bf lo = *(const v8bf*)(p);
  v8bf hi = *(const v8bf*)(p + 16);
  v16bf r;
#pragma unroll
  for (int i = 0; i < 8; ++i) { r[i] = lo[i]; r[i + 8] = hi[i]; }
  return r;
}

// Same A fragment, converting from fp32 source on the fly.
__device__ __forceinline__ v16bf load_a_frag_f32(const float* p0, int stride, int lane) {
  const int m = lane & 15, g = lane >> 4;
  const float* p = p0 + (size_t)m * stride + g * 8;
  v16bf r;
#pragma unroll
  for (int i = 0; i < 8; ++i) r[i] = (bf16_t)p[i];
#pragma unroll
  for (int i = 0; i < 8; ++i) r[i + 8] = (bf16_t)p[16 + i];
  return r;
}

// B-matrix 32x16 bf16 fragment from row-major Mat[n][k] (n = output column):
// lane n=lane&15, half g=lane>>4, elems 0..7 <- k = g*8 + 0..7 ; elems 8..15 <- k = 16 + g*8 + 0..7
__device__ __forceinline__ v16bf load_b_frag(const bf16_t* p0, int stride, int lane) {
  const int n = lane & 15, g = lane >> 4;
  const bf16_t* p = p0 + (size_t)n * stride + g * 8;
  v8bf lo = *(const v8bf*)(p);
  v8bf hi = *(const v8bf*)(p + 16);
  v16bf r;
#pragma unroll
  for (int i = 0; i < 8; ++i) { r[i] = lo[i]; r[i + 8] = hi[i]; }
  return r;
}

// ---------------------------------------------------------------------------
// Kernel 1: elementwise conversions to bf16 (V = x, Q = tanh(x'), Wb = W)
// ---------------------------------------------------------------------------
__global__ void convert_kernel(const float* __restrict__ x,
                               const float* __restrict__ xp,
                               const float* __restrict__ W,
                               bf16_t* __restrict__ Vb,
                               bf16_t* __restrict__ Qt,
                               bf16_t* __restrict__ Wb) {
  const long long N  = (long long)BATCH * SEQ * CH;
  const long long NW = (long long)CH * CH;
  const long long step = (long long)gridDim.x * blockDim.x;
  for (long long i = (long long)blockIdx.x * blockDim.x + threadIdx.x; i < N; i += step) {
    Vb[i] = (bf16_t)x[i];
    Qt[i] = (bf16_t)tanhf(xp[i]);
    if (i < NW) Wb[i] = (bf16_t)W[i];
  }
}

// ---------------------------------------------------------------------------
// Kernel 2: K = bf16(x' @ W^T + b).  Grid: (rows/64, C/128), 4 waves/WG.
// W chunk (128 cols x 512 k, 128 KB) staged once in LDS via async copies and
// shared by all 4 waves; A-fragments converted f32->bf16 on the fly.
// ---------------------------------------------------------------------------
__global__ __launch_bounds__(128) void linear_kernel(
    const float*  __restrict__ xp,    // [B*S, C] f32
    const bf16_t* __restrict__ Wb,    // [C, C]  bf16, row d, col c
    const float*  __restrict__ bias,  // [C]
    bf16_t* __restrict__ Kq) {        // [B*S, C] bf16
  __shared__ __align__(16) bf16_t Ws[128][CH];   // 128 KB

  const int tid  = threadIdx.x;
  const int lane = tid & 31;
  const int wave = tid >> 5;
  const int mrow = blockIdx.x * 64 + wave * 16;
  const int c0   = blockIdx.y * 128;

  {
    const char* Wg = (const char*)(Wb + (size_t)c0 * CH);
    char* Wl = (char*)&Ws[0][0];
    for (int i = tid; i < 128 * CH * 2 / 16; i += 128)
      async_copy_b128(Wl + i * 16, Wg + i * 16);
  }

  v8f acc[8];
#pragma unroll
  for (int ct = 0; ct < 8; ++ct)
#pragma unroll
    for (int i = 0; i < 8; ++i) acc[ct][i] = 0.0f;

  wait_async0();
  __syncthreads();

  const float* arow = xp + (size_t)mrow * CH;
#pragma unroll 4
  for (int k = 0; k < CH; k += 32) {
    v16bf a = load_a_frag_f32(arow + k, CH, lane);
#pragma unroll
    for (int ct = 0; ct < 8; ++ct) {
      v16bf b = load_b_frag(&Ws[ct * 16][k], CH, lane);
      acc[ct] = wmma_bf16(a, b, acc[ct]);
    }
  }

  const int g = lane >> 4, n = lane & 15;
#pragma unroll
  for (int ct = 0; ct < 8; ++ct) {
    const int col = c0 + ct * 16 + n;
    const float bv = bias[col];
#pragma unroll
    for (int r = 0; r < 8; ++r) {
      const int row = mrow + r + 8 * g;
      Kq[(size_t)row * CH + col] = (bf16_t)(acc[ct][r] + bv);
    }
  }
}

// ---------------------------------------------------------------------------
// Kernel 3: fused flash attention + final tanh, software-pipelined.
// Grid: (S/BM, B), 4 waves.  LDS = Q(64K) + K double-buffer(128K) + Vt(64K)
// + alpha(8K) = 264 KB -> 1 WG/WGP, so the K tile for iteration t+1 is
// async-prefetched under iteration t's compute (only waited at iter end),
// and the V fill runs after QK issue so its global loads hide under WMMA.
// ---------------------------------------------------------------------------
__global__ __launch_bounds__(128, 1) void flash_kernel(
    const bf16_t* __restrict__ Qt,   // tanh(x') bf16 [B*S, C]
    const bf16_t* __restrict__ Kq,   // linear out bf16 [B*S, C]
    const bf16_t* __restrict__ V,    // x bf16 [B*S, C]
    float* __restrict__ out) {       // [B*S, C] f32
  __shared__ __align__(16) bf16_t Qs[BM][CH];        // 64 KB
  __shared__ __align__(16) bf16_t Ks[2][BN][CH];     // 128 KB (double buffer)
  __shared__ __align__(16) bf16_t Vts[CH][BN];       // 64 KB (transposed)
  __shared__ __align__(16) bf16_t Alphas[BM][BN];    //  8 KB

  const int b    = blockIdx.y;
  const int m0   = blockIdx.x * BM;
  const int tid  = threadIdx.x;
  const int lane = tid & 31;
  const int wave = tid >> 5;
  const int mw   = wave * 16;
  const int g    = lane >> 4, n = lane & 15;

  const bf16_t* Kbase = Kq + (size_t)b * SEQ * CH;
  const bf16_t* Vbase = V  + (size_t)b * SEQ * CH;

  // Prologue: async-stage Q tile and K tile 0.
  {
    const char* Qg = (const char*)(Qt + ((size_t)b * SEQ + m0) * CH);
    char* Ql = (char*)&Qs[0][0];
    for (int i = tid; i < BM * CH * 2 / 16; i += 128)
      async_copy_b128(Ql + i * 16, Qg + i * 16);
    const char* Kg = (const char*)Kbase;
    char* Kl = (char*)&Ks[0][0][0];
    for (int i = tid; i < BN * CH * 2 / 16; i += 128)
      async_copy_b128(Kl + i * 16, Kg + i * 16);
  }

  float M[8], L[8];
#pragma unroll
  for (int r = 0; r < 8; ++r) { M[r] = -3.0e38f; L[r] = 0.0f; }
  v8f O[32];
#pragma unroll
  for (int ct = 0; ct < 32; ++ct)
#pragma unroll
    for (int i = 0; i < 8; ++i) O[ct][i] = 0.0f;

  wait_async0();

  for (int it = 0; it < SEQ / BN; ++it) {
    const int t0  = it * BN;
    const int cur = it & 1;

    __syncthreads();  // prev PV done (Vts reusable), K[cur] visible to all

    // Prefetch next K tile into the other buffer; hidden under this iter.
    if (t0 + BN < SEQ) {
      const char* Kg = (const char*)(Kbase + (size_t)(t0 + BN) * CH);
      char* Kl = (char*)&Ks[cur ^ 1][0][0];
      for (int i = tid; i < BN * CH * 2 / 16; i += 128)
        async_copy_b128(Kl + i * 16, Kg + i * 16);
    }

    // ---- scores: S = Q(16x512) . K^T(512x64) -> 4 tiles of 16x16 ----
    v8f Sc[4];
#pragma unroll
    for (int ct = 0; ct < 4; ++ct)
#pragma unroll
      for (int i = 0; i < 8; ++i) Sc[ct][i] = 0.0f;
#pragma unroll 4
    for (int k = 0; k < CH; k += 32) {
      v16bf a = load_a_frag(&Qs[mw][k], CH, lane);
#pragma unroll
      for (int ct = 0; ct < 4; ++ct) {
        v16bf kb = load_b_frag(&Ks[cur][ct * 16][k], CH, lane);
        Sc[ct] = wmma_bf16(a, kb, Sc[ct]);
      }
    }

    // ---- V tile fill (transpose via VGPRs); QK above hides the latency ----
    {
      const bf16_t* Vg = Vbase + (size_t)t0 * CH;
      for (int i = tid; i < BN * CH / 8; i += 128) {
        const int t = (i * 8) / CH, c = (i * 8) % CH;
        v8bf d = ((const v8bf*)Vg)[i];
#pragma unroll
        for (int j = 0; j < 8; ++j) Vts[c + j][t] = d[j];
      }
    }

    // ---- online softmax (rows r+8g live in this lane's half-wave) ----
    float Mnew[8], scale[8], psum[8];
#pragma unroll
    for (int r = 0; r < 8; ++r) {
      float t = Sc[0][r];
#pragma unroll
      for (int ct = 1; ct < 4; ++ct) t = fmaxf(t, Sc[ct][r]);
#pragma unroll
      for (int m = 1; m < 16; m <<= 1) t = fmaxf(t, __shfl_xor(t, m, 32));
      Mnew[r]  = fmaxf(M[r], t);
      scale[r] = __expf(M[r] - Mnew[r]);
      psum[r]  = 0.0f;
    }
#pragma unroll
    for (int ct = 0; ct < 4; ++ct)
#pragma unroll
      for (int r = 0; r < 8; ++r) {
        float p = __expf(Sc[ct][r] - Mnew[r]);
        Sc[ct][r] = p;
        psum[r] += p;
      }
#pragma unroll
    for (int r = 0; r < 8; ++r) {
#pragma unroll
      for (int m = 1; m < 16; m <<= 1) psum[r] += __shfl_xor(psum[r], m, 32);
      L[r] = L[r] * scale[r] + psum[r];
      M[r] = Mnew[r];
    }
    // rescale running output
#pragma unroll
    for (int ct = 0; ct < 32; ++ct)
#pragma unroll
      for (int r = 0; r < 8; ++r) O[ct][r] *= scale[r];

    // ---- re-layout alpha D-frag -> A-frag via LDS (own band only) ----
#pragma unroll
    for (int ct = 0; ct < 4; ++ct)
#pragma unroll
      for (int r = 0; r < 8; ++r)
        Alphas[mw + r + 8 * g][ct * 16 + n] = (bf16_t)Sc[ct][r];

    __syncthreads();  // Vts complete across all waves before PV

    // ---- O += alpha(16x64) . V(64x512) ----
#pragma unroll
    for (int ks = 0; ks < BN; ks += 32) {
      v16bf a = load_a_frag(&Alphas[mw][ks], BN, lane);
#pragma unroll
      for (int ct = 0; ct < 32; ++ct) {
        v16bf vb = load_b_frag(&Vts[ct * 16][ks], BN, lane);
        O[ct] = wmma_bf16(a, vb, O[ct]);
      }
    }

    wait_async0();  // K[next] in LDS before the top-of-loop barrier
  }

  // ---- epilogue: normalize, tanh, store fp32 ----
  float Linv[8];
#pragma unroll
  for (int r = 0; r < 8; ++r) Linv[r] = 1.0f / L[r];
#pragma unroll
  for (int ct = 0; ct < 32; ++ct)
#pragma unroll
    for (int r = 0; r < 8; ++r) {
      const int row = m0 + mw + r + 8 * g;
      const int col = ct * 16 + n;
      out[((size_t)b * SEQ + row) * CH + col] = tanhf(O[ct][r] * Linv[r]);
    }
}

// ---------------------------------------------------------------------------
extern "C" void kernel_launch(void* const* d_in, const int* in_sizes, int n_in,
                              void* d_out, int out_size, void* d_ws, size_t ws_size,
                              hipStream_t stream) {
  const float* x    = (const float*)d_in[0];
  const float* xp   = (const float*)d_in[1];
  const float* W    = (const float*)d_in[2];
  const float* bias = (const float*)d_in[3];
  float* out        = (float*)d_out;

  char* ws = (char*)d_ws;
  const size_t SZ = (size_t)BATCH * SEQ * CH * sizeof(bf16_t);  // 32 MiB
  bf16_t* Kq = (bf16_t*)(ws);
  bf16_t* Qt = (bf16_t*)(ws + SZ);
  bf16_t* Vb = (bf16_t*)(ws + 2 * SZ);
  bf16_t* Wb = (bf16_t*)(ws + 3 * SZ);

  convert_kernel<<<8192, 256, 0, stream>>>(x, xp, W, Vb, Qt, Wb);
  linear_kernel<<<dim3((BATCH * SEQ) / 64, CH / 128), 128, 0, stream>>>(xp, Wb, bias, Kq);
  flash_kernel<<<dim3(SEQ / BM, BATCH), 128, 0, stream>>>(Qt, Kq, Vb, out);
}
